// DenseAnyLocGcn_41755672051730
// MI455X (gfx1250) — matrix operand order repaired and checked by reading, not compiled, not run on hardware
//
#include <hip/hip_runtime.h>
#include <hip/hip_bf16.h>
#include <math.h>

typedef __attribute__((ext_vector_type(16))) _Float16 v16h;
typedef __attribute__((ext_vector_type(8)))  float    v8f;

#define TPB 256

static const int Nn = 19000;    // nodes
static const int En = 304000;   // edges
static const int Gg = 1000;     // graphs (N/19)
static const int CH = 19000;    // featconv row chunk (N = 1 chunk, E = 16 chunks)

static inline int gridFor(int n) { return (n + TPB - 1) / TPB; }
static inline int imin(int a, int b) { return a < b ? a : b; }

__device__ __forceinline__ int dmin(int a, int b) { return a < b ? a : b; }
__device__ __forceinline__ int dmax(int a, int b) { return a > b ? a : b; }

// ---------------------------------------------------------------------------
// WMMA GEMM: C(MxN) = act(A(MxK) @ B(KxN) + bias), f32 in/out, f16 multiply
// via v_wmma_f32_16x16x32_f16.
//
// Block (8 waves) computes a 256(M) x 32(N) tile; wave w owns a 32x32 tile
// (2x2 WMMAs). The 32(K)x32(N) B tile is shared by all 8 waves, so it is
// loaded cooperatively ONCE per K-step, converted to f16 (v_cvt_pk_f16_f32)
// and staged in LDS already in WMMA B-operand layout (each lane then reads
// its 16 halves as two ds_load_b128). Double-buffered: one __syncthreads per
// K-step; staging of step s+1 overlaps the 4 WMMAs of step s. Staging is
// gather-then-scatter: all global loads issue first (one clause), then
// packed b32 LDS stores.
//
// Bounds strategy (branchless): M/N overruns only affect output rows/cols
// that the store guards discard, so row/col indices are merely CLAMPED (no
// masking, no EXEC divergence; waves never early-exit so barriers are
// uniform). K overruns contaminate every output, so A uses a mask-free main
// body (uniform branch) + masked last step; B staging masks K always.
//
// CONV=1: A is generated on the fly as the 9-tap im2col of X (Lfull x cin):
//   A[m, k] = X[rowOff + m + (k/cin) - 4, k%cin]  (zero outside [0,Lfull))
// ---------------------------------------------------------------------------
template<int CONV>
__global__ void wmma_gemm(const float* __restrict__ A, int lda,
                          const float* __restrict__ B, int ldb,
                          const float* __restrict__ bias,
                          float* __restrict__ C, int ldc,
                          int M, int N, int K, int relu,
                          int cin, int Lfull, int rowOff)
{
    __shared__ _Float16 Bs[2][1024];    // 2 x (32K x 32N) f16 tiles, operand layout

    const int tid  = threadIdx.x;
    const int lane = tid & 31;
    const int wv   = tid >> 5;          // wave in block: 0..7
    const int tilesN = (N + 31) >> 5;
    const int bm = blockIdx.x / tilesN; // 256-row block tile index
    const int tn = blockIdx.x - bm * tilesN;
    const int ln = lane & 15;
    const int hi = lane >> 4;
    const int mbase = bm * 256 + wv * 32;

    int mcl[2], ncol[2], ncl[2];
#pragma unroll
    for (int i = 0; i < 2; ++i) {
        mcl[i]  = dmin(mbase + i * 16 + ln, M - 1);  // clamped A row (garbage rows never stored)
        ncol[i] = tn * 32 + i * 16 + ln;             // C col for this lane
        ncl[i]  = dmin(ncol[i], N - 1);
    }

    v8f acc[2][2];
#pragma unroll
    for (int ni = 0; ni < 2; ++ni) {
        const float bv = bias[ncl[ni]];
#pragma unroll
        for (int mi = 0; mi < 2; ++mi)
#pragma unroll
            for (int r = 0; r < 8; ++r) acc[mi][ni][r] = bv;
    }

    // Thread handles 2 K-pairs: pair pe covers elements (kk, nn2), (kk+1, nn2)
    // with kk even, so both land in lane l = ((kk>>3)&1)*16 + (nn2&15), group
    // ni = nn2>>4, at adjacent half-slots ((kk&7)+8*(kk>>4), +1) -> one b32 store.
    const int pe0  = tid;                 // pairs tid and tid+256 (512 total)
    auto stageB = [&](int k0, int buf) {
        float v[2][2];
        // gather phase: all 4 global loads issue together
#pragma unroll
        for (int q = 0; q < 2; ++q) {
            const int pe  = pe0 + q * 256;
            const int kk  = (pe >> 5) << 1;    // 0,2,..,30
            const int nn2 = pe & 31;
            const int nc  = dmin(tn * 32 + nn2, N - 1);
#pragma unroll
            for (int p = 0; p < 2; ++p) {
                const int kg = k0 + kk + p;
                const float bw = B[(long)dmin(kg, K - 1) * ldb + nc];
                v[q][p] = (kg < K) ? bw : 0.0f; // K overrun contaminates; N overrun harmless
            }
        }
        // scatter phase: pack 2 halves -> one aligned b32 LDS store
#pragma unroll
        for (int q = 0; q < 2; ++q) {
            const int pe  = pe0 + q * 256;
            const int kk  = (pe >> 5) << 1;
            const int nn2 = pe & 31;
            const int l   = (((kk >> 3) & 1) << 4) | (nn2 & 15);
            const int idx = ((nn2 >> 4) * 32 + l) * 16 + ((kk & 7) | ((kk >> 4) << 3));   // even -> 4B aligned
            union { _Float16 h[2]; unsigned int u; } pk;
            pk.h[0] = (_Float16)v[q][0];
            pk.h[1] = (_Float16)v[q][1];
            *(unsigned int*)&Bs[buf][idx] = pk.u;
        }
    };

    const int steps = (K + 31) >> 5;
    stageB(0, 0);

    for (int s = 0; s < steps; ++s) {
        const int k0 = s << 5;
        __syncthreads();                              // Bs[s&1] ready for all waves
        if (s + 1 < steps) stageB((s + 1) << 5, (s + 1) & 1);

        v16h a[2], b[2];
        b[0] = *(const v16h*)&Bs[s & 1][(lane) * 16];
        b[1] = *(const v16h*)&Bs[s & 1][(32 + lane) * 16];

        if (k0 + 32 <= K) {
            // ---- mask-free A loads ----
#pragma unroll
            for (int j = 0; j < 8; ++j) {
                // 16-bit A 16x32 layout: V0..3 -> K 0..7 (+8 lanes 16-31), V4..7 -> K 16..23 (+8)
                const int kaBase = (j < 4) ? (2 * j + 8 * hi) : (16 + 2 * (j - 4) + 8 * hi);
#pragma unroll
                for (int mi = 0; mi < 2; ++mi) {
                    if (CONV) {
#pragma unroll
                        for (int p = 0; p < 2; ++p) {
                            const int ka  = k0 + kaBase + p;
                            const int t   = ka / cin;
                            const int c   = ka - t * cin;
                            const int src = rowOff + mcl[mi] + t - 4;
                            const int scl = dmax(0, dmin(src, Lfull - 1));
                            float av = A[(long)scl * cin + c];       // unconditional load
                            if (src < 0 || src >= Lfull) av = 0.0f;  // conv zero-padding
                            a[mi][2 * j + p] = (_Float16)av;
                        }
                    } else {
                        // pair (k, k+1) contiguous -> single b64 load (all ldas even)
                        const float2 v2 = *(const float2*)&A[(long)mcl[mi] * lda + k0 + kaBase];
                        a[mi][2 * j]     = (_Float16)v2.x;
                        a[mi][2 * j + 1] = (_Float16)v2.y;
                    }
                }
            }
        } else {
            // ---- masked A tail (last step only) ----
#pragma unroll
            for (int j = 0; j < 8; ++j) {
                const int kaBase = (j < 4) ? (2 * j + 8 * hi) : (16 + 2 * (j - 4) + 8 * hi);
#pragma unroll
                for (int p = 0; p < 2; ++p) {
                    const int ka   = k0 + kaBase + p;
                    const bool okA = ka < K;
                    const int kac  = okA ? ka : (K - 1);
#pragma unroll
                    for (int mi = 0; mi < 2; ++mi) {
                        float av;
                        if (CONV) {
                            const int t   = kac / cin;
                            const int c   = kac - t * cin;
                            const int src = rowOff + mcl[mi] + t - 4;
                            const int scl = dmax(0, dmin(src, Lfull - 1));
                            av = A[(long)scl * cin + c];
                            if (src < 0 || src >= Lfull || !okA) av = 0.0f;
                        } else {
                            av = A[(long)mcl[mi] * lda + kac];
                            if (!okA) av = 0.0f;
                        }
                        a[mi][2 * j + p] = (_Float16)av;
                    }
                }
            }
        }

#pragma unroll
        for (int mi = 0; mi < 2; ++mi)
#pragma unroll
            for (int ni = 0; ni < 2; ++ni)
                acc[mi][ni] = __builtin_amdgcn_wmma_f32_16x16x32_f16(
                    false, a[mi], false, b[ni], (short)0, acc[mi][ni], false, false);
    }

    // ---- store (C/D 16x16 f32 layout: VGPR r holds row r (+8 for lanes 16-31)) ----
#pragma unroll
    for (int mi = 0; mi < 2; ++mi)
#pragma unroll
        for (int ni = 0; ni < 2; ++ni)
#pragma unroll
            for (int r = 0; r < 8; ++r) {
                const int m = mbase + mi * 16 + r + 8 * hi;
                const int n = ncol[ni];
                if (m < M && n < N) {
                    float v = acc[mi][ni][r];
                    if (relu) v = v > 0.0f ? v : 0.0f;
                    C[(long)m * ldc + n] = v;
                }
            }
}

// ---------------------------------------------------------------------------
// Elementwise / graph kernels
// ---------------------------------------------------------------------------
__global__ void k_fill(float* p, float v, int n) {
    int i = blockIdx.x * blockDim.x + threadIdx.x;
    if (i < n) p[i] = v;
}

__global__ void k_deg(const int* __restrict__ col, float* deg, int E) {
    int e = blockIdx.x * blockDim.x + threadIdx.x;
    if (e < E) atomicAdd(&deg[col[e]], 1.0f);
}

__global__ void k_dis(float* deg, int n) {   // deg -> deg>0 ? rsqrt(deg) : 0  (in place)
    int i = blockIdx.x * blockDim.x + threadIdx.x;
    if (i < n) { float d = deg[i]; deg[i] = d > 0.0f ? rsqrtf(d) : 0.0f; }
}

__global__ void k_ew(const int* __restrict__ row, const int* __restrict__ col,
                     const float* __restrict__ dis, float* ew, int E) {
    int e = blockIdx.x * blockDim.x + threadIdx.x;
    if (e < E) ew[e] = dis[row[e]] * dis[col[e]];
}

// Scatter conv weights w_i (OC, cin, k=2i+1) into combined Wc (9*cin, 500)
__global__ void k_repack_w(const float* __restrict__ w, float* __restrict__ wc,
                           int i, int cin, int total) {
    int idx = blockIdx.x * blockDim.x + threadIdx.x;
    if (idx >= total) return;
    const int k  = 2 * i + 1;
    const int oc = idx / (cin * k);
    const int rm = idx - oc * cin * k;
    const int c  = rm / k;
    const int dk = rm - c * k;
    const int rowi = (4 - i + dk) * cin + c;   // tap dk of conv i sits at window offset 4-i+dk
    wc[rowi * 500 + i * 100 + oc] = w[idx];
}

__global__ void k_repack_b(const float* b0, const float* b1, const float* b2,
                           const float* b3, const float* b4, float* bc) {
    int j = blockIdx.x * blockDim.x + threadIdx.x;
    if (j >= 500) return;
    const float* bs[5] = {b0, b1, b2, b3, b4};
    bc[j] = bs[j / 100][j % 100];
}

__global__ void k_concat(const float* __restrict__ a, int ca,
                         const float* __restrict__ b, int cb,
                         float* __restrict__ dst, int rows) {
    int idx = blockIdx.x * blockDim.x + threadIdx.x;
    const int cw = ca + cb;
    if (idx >= rows * cw) return;
    const int i = idx / cw, j = idx - i * cw;
    dst[idx] = (j < ca) ? a[i * ca + j] : b[i * cb + (j - ca)];
}

// gate[e] = sum_j ((x[row]+e)[j]*ew) * attW[j] + attb   (msg recomputed on the fly)
__global__ void k_gate(const float* __restrict__ xf, const float* __restrict__ ef,
                       const float* __restrict__ ew, const int* __restrict__ row,
                       const float* __restrict__ attW, const float* __restrict__ attb,
                       float* __restrict__ gate, int E) {
    int e = blockIdx.x * blockDim.x + threadIdx.x;
    if (e >= E) return;
    const int r = row[e];
    const float w = ew[e];
    float g = 0.0f;
#pragma unroll 4
    for (int j = 0; j < 100; ++j)
        g += (xf[r * 100 + j] + ef[e * 100 + j]) * w * attW[j];
    gate[e] = g + attb[0];
}

__device__ inline void atomicMaxF(float* addr, float val) {
    unsigned int* u = (unsigned int*)addr;
    unsigned int old = __float_as_uint(*addr);
    while (__uint_as_float(old) < val) {
        unsigned int prev = atomicCAS(u, old, __float_as_uint(val));
        if (prev == old) break;
        old = prev;
    }
}

__global__ void k_gmax(const float* __restrict__ gate, const int* __restrict__ col,
                       float* gmax, int E) {
    int e = blockIdx.x * blockDim.x + threadIdx.x;
    if (e < E) atomicMaxF(&gmax[col[e]], gate[e]);
}

__global__ void k_gmax_fix(float* gmax, int n) {   // where(isfinite(gmax), gmax, 0)
    int i = blockIdx.x * blockDim.x + threadIdx.x;
    if (i < n && gmax[i] < -1e29f) gmax[i] = 0.0f;
}

__global__ void k_exden(const float* __restrict__ gate, const float* __restrict__ gmax,
                        const int* __restrict__ col, float* __restrict__ ex,
                        float* __restrict__ den, int E) {
    int e = blockIdx.x * blockDim.x + threadIdx.x;
    if (e >= E) return;
    const int c = col[e];
    const float v = expf(gate[e] - gmax[c]);
    ex[e] = v;
    atomicAdd(&den[c], v);
}

// one thread per (edge, channel): aggr[col, j] += alpha * msg[e, j]
__global__ void k_aggr(const float* __restrict__ xf, const float* __restrict__ ef,
                       const float* __restrict__ ew, const float* __restrict__ ex,
                       const float* __restrict__ den, const int* __restrict__ row,
                       const int* __restrict__ col, float* __restrict__ aggr, int E) {
    int idx = blockIdx.x * blockDim.x + threadIdx.x;
    if (idx >= E * 100) return;
    const int e = idx / 100, j = idx - e * 100;
    const int c = col[e];
    const float alpha = ex[e] / (den[c] + 1e-16f);
    const float msg = (xf[row[e] * 100 + j] + ef[e * 100 + j]) * ew[e];
    atomicAdd(&aggr[c * 100 + j], alpha * msg);
}

__global__ void k_gru(const float* __restrict__ gi, const float* __restrict__ gh,
                      const float* __restrict__ x, float* __restrict__ out, int n) {
    int idx = blockIdx.x * blockDim.x + threadIdx.x;
    if (idx >= n) return;
    const int i = idx / 100, c = idx - i * 100;
    const float ir = gi[i * 300 + c],       hr = gh[i * 300 + c];
    const float iz = gi[i * 300 + 100 + c], hz = gh[i * 300 + 100 + c];
    const float in = gi[i * 300 + 200 + c], hn = gh[i * 300 + 200 + c];
    const float r = 1.0f / (1.0f + expf(-(ir + hr)));
    const float z = 1.0f / (1.0f + expf(-(iz + hz)));
    const float nn = tanhf(in + r * hn);
    out[idx] = (1.0f - z) * nn + z * x[idx];
}

// ---------------------------------------------------------------------------
// Parameter pytree mapping (JAX alphabetical flatten within each dict)
// ---------------------------------------------------------------------------
struct FCp { const float* b[5]; const float *dW1,*dW2,*dW3,*db1,*db2,*db3; const float* w[5]; };
struct Lp  { const float *Wh,*Wi,*attW,*attb,*bh,*bi; FCp edge, node; };

static FCp getFC(void* const* din, int base) {
    FCp f;
    for (int i = 0; i < 5; ++i) f.b[i] = (const float*)din[base + i];
    f.dW1 = (const float*)din[base + 5];
    f.dW2 = (const float*)din[base + 6];
    f.dW3 = (const float*)din[base + 7];
    f.db1 = (const float*)din[base + 8];
    f.db2 = (const float*)din[base + 9];
    f.db3 = (const float*)din[base + 10];
    for (int i = 0; i < 5; ++i) f.w[i] = (const float*)din[base + 11 + i];
    return f;
}
static Lp getLayer(void* const* din, int base) {
    Lp L;
    L.Wh   = (const float*)din[base + 0];
    L.Wi   = (const float*)din[base + 1];
    L.attW = (const float*)din[base + 2];
    L.attb = (const float*)din[base + 3];
    L.bh   = (const float*)din[base + 4];
    L.bi   = (const float*)din[base + 5];
    L.edge = getFC(din, base + 6);
    L.node = getFC(din, base + 22);
    return L;
}

// ---------------------------------------------------------------------------
extern "C" void kernel_launch(void* const* d_in, const int* in_sizes, int n_in,
                              void* d_out, int out_size, void* d_ws, size_t ws_size,
                              hipStream_t stream)
{
    (void)out_size; (void)ws_size;
    // Detect top-level ordering via unambiguous sizes:
    //   alphabetical flatten: edge_attr(1216000), edge_index(608000), params..., x(76000)
    //   insertion order:      x(76000), edge_attr(1216000), params..., edge_index(608000)
    int idx_ea, idx_ei, idx_x;
    const int pbase = 2;
    if (n_in >= 85 && in_sizes[0] == 1216000 && in_sizes[1] == 608000) {
        idx_ea = 0; idx_ei = 1; idx_x = 84;
    } else {
        idx_x = 0; idx_ea = 1; idx_ei = 84;
    }
    const float* x         = (const float*)d_in[idx_x];
    const float* edge_attr = (const float*)d_in[idx_ea];
    const int*   ei        = (const int*)d_in[idx_ei];
    const int*   row       = ei;
    const int*   col       = ei + En;

    const Lp L1 = getLayer(d_in, pbase);        // layer1 at 2..39
    const Lp L2 = getLayer(d_in, pbase + 38);   // layer2 at 40..77
    const float* mW1 = (const float*)d_in[pbase + 76];
    const float* mW2 = (const float*)d_in[pbase + 77];
    const float* mW3 = (const float*)d_in[pbase + 78];
    const float* mb1 = (const float*)d_in[pbase + 79];
    const float* mb2 = (const float*)d_in[pbase + 80];
    const float* mb3 = (const float*)d_in[pbase + 81];

    // ---- workspace carve-up (f32, 256B aligned), ~390 MB total ----
    char* wp = (char*)d_ws;
    auto alloc = [&](size_t elems) {
        float* r = (float*)wp;
        wp += ((elems * sizeof(float) + 255) & ~(size_t)255);
        return r;
    };
    float* deg  = alloc(Nn);              // degree -> dis (in place)
    float* ew   = alloc(En);
    float* wc   = alloc(936 * 500);       // combined conv weights (max 9*104 x 500)
    float* bc   = alloc(500);
    float* hbuf = alloc((size_t)CH * 500);
    float* h1   = alloc((size_t)CH * 64);
    float* h2   = alloc((size_t)CH * 32);
    float* xf1  = alloc((size_t)Nn * 100);
    float* x1   = alloc((size_t)Nn * 100);
    float* x1c  = alloc((size_t)Nn * 104);
    float* e1   = alloc((size_t)En * 100);
    float* e1c  = alloc((size_t)En * 104);
    float* gate = alloc(En);
    float* exb  = alloc(En);
    float* gmax = alloc(Nn);
    float* den  = alloc(Nn);
    float* aggr = alloc((size_t)Nn * 100);
    float* gi   = alloc((size_t)Nn * 300);
    float* gh   = alloc((size_t)Nn * 300);
    float* xf2  = alloc((size_t)Nn * 100);
    float* x2   = alloc((size_t)Nn * 100);
    float* hA   = alloc((size_t)Gg * 512);
    float* hB   = alloc((size_t)Gg * 64);
    float* e2   = e1;   // e1 is dead once e1c is built -> reuse for layer2 edge features

    auto gemmBlocks = [](int M, int Nc) {
        return ((M + 255) / 256) * ((Nc + 31) / 32);
    };
    auto gemm = [&](const float* A, int lda, const float* B, int ldb, const float* bias,
                    float* C, int ldc, int M, int Nc, int K, int relu) {
        wmma_gemm<0><<<gemmBlocks(M, Nc), TPB, 0, stream>>>(
            A, lda, B, ldb, bias, C, ldc, M, Nc, K, relu, 0, 0, 0);
    };

    auto featconv = [&](const float* X, int L, int cin, const FCp& p, float* out) {
        const int Kc = 9 * cin;
        k_fill<<<gridFor(Kc * 500), TPB, 0, stream>>>(wc, 0.0f, Kc * 500);
        for (int i = 0; i < 5; ++i) {
            const int kk = 2 * i + 1, total = 100 * cin * kk;
            k_repack_w<<<gridFor(total), TPB, 0, stream>>>(p.w[i], wc, i, cin, total);
        }
        k_repack_b<<<gridFor(500), TPB, 0, stream>>>(p.b[0], p.b[1], p.b[2], p.b[3], p.b[4], bc);
        for (int s0 = 0; s0 < L; s0 += CH) {
            const int CL = imin(CH, L - s0);
            // fused im2col9 + GEMM: hbuf = im2col(X[s0..]) @ Wc + bc
            wmma_gemm<1><<<gemmBlocks(CL, 500), TPB, 0, stream>>>(
                X, 0, wc, 500, bc, hbuf, 500, CL, 500, Kc, 0, cin, L, s0);
            gemm(hbuf, 500, p.dW1, 64,  p.db1, h1, 64,  CL, 64,  500, 1);
            gemm(h1,   64,  p.dW2, 32,  p.db2, h2, 32,  CL, 32,  64,  1);
            gemm(h2,   32,  p.dW3, 100, p.db3, out + (long)s0 * 100, 100, CL, 100, 32, 0);
        }
    };

    auto run_layer = [&](const float* xin, int cinN, const float* ein, int cinE,
                         const Lp& P, float* xfeat, float* efeat, float* xout) {
        featconv(xin, Nn, cinN, P.node, xfeat);
        featconv(ein, En, cinE, P.edge, efeat);
        k_gate<<<gridFor(En), TPB, 0, stream>>>(xfeat, efeat, ew, row, P.attW, P.attb, gate, En);
        k_fill<<<gridFor(Nn), TPB, 0, stream>>>(gmax, -1e30f, Nn);
        k_gmax<<<gridFor(En), TPB, 0, stream>>>(gate, col, gmax, En);
        k_gmax_fix<<<gridFor(Nn), TPB, 0, stream>>>(gmax, Nn);
        k_fill<<<gridFor(Nn), TPB, 0, stream>>>(den, 0.0f, Nn);
        k_exden<<<gridFor(En), TPB, 0, stream>>>(gate, gmax, col, exb, den, En);
        k_fill<<<gridFor(Nn * 100), TPB, 0, stream>>>(aggr, 0.0f, Nn * 100);
        k_aggr<<<gridFor(En * 100), TPB, 0, stream>>>(xfeat, efeat, ew, exb, den, row, col, aggr, En);
        gemm(aggr,  100, P.Wi, 300, P.bi, gi, 300, Nn, 300, 100, 0);
        gemm(xfeat, 100, P.Wh, 300, P.bh, gh, 300, Nn, 300, 100, 0);
        k_gru<<<gridFor(Nn * 100), TPB, 0, stream>>>(gi, gh, xfeat, xout, Nn * 100);
    };

    // ---- degree / symmetric normalization / edge weights ----
    k_fill<<<gridFor(Nn), TPB, 0, stream>>>(deg, 0.0f, Nn);
    k_deg<<<gridFor(En), TPB, 0, stream>>>(col, deg, En);
    k_dis<<<gridFor(Nn), TPB, 0, stream>>>(deg, Nn);
    k_ew<<<gridFor(En), TPB, 0, stream>>>(row, col, deg, ew, En);

    // ---- layer 1 (cin = 4) ----
    run_layer(x, 4, edge_attr, 4, L1, xf1, e1, x1);
    k_concat<<<gridFor(Nn * 104), TPB, 0, stream>>>(x1, 100, x,         4, x1c, Nn);
    k_concat<<<gridFor(En * 104), TPB, 0, stream>>>(e1, 100, edge_attr, 4, e1c, En);

    // ---- layer 2 (cin = 104) ----
    run_layer(x1c, 104, e1c, 104, L2, xf2, e2, x2);

    // ---- MLP head: (1000, 1900) -> 512 -> 64 -> 34 ----
    gemm(x2, 1900, mW1, 512, mb1, hA, 512, Gg, 512, 1900, 1);
    gemm(hA, 512,  mW2, 64,  mb2, hB, 64,  Gg, 64,  512,  1);
    gemm(hB, 64,   mW3, 34,  mb3, (float*)d_out, 34, Gg, 34, 64, 0);
}
